// DPINeuron_55448027792027
// MI455X (gfx1250) — matrix operand order repaired and checked by reading, not compiled, not run on hardware
//
#include <hip/hip_runtime.h>
#include <hip/hip_bf16.h>

typedef __attribute__((ext_vector_type(16))) _Float16 v16h;
typedef __attribute__((ext_vector_type(8)))  _Float16 v8h;
typedef __attribute__((ext_vector_type(4)))  _Float16 v4h;
typedef __attribute__((ext_vector_type(8)))  float    v8f;

#define B_DIM 4096
#define N_IN  2048
#define N_OUT 2048
#define LDX   40   // padded LDS row stride (halves): 80B, 16B-aligned, conflict-free

// Build a 16-half WMMA fragment from two contiguous 16-byte LDS chunks.
__device__ __forceinline__ v16h ld_frag(const _Float16* p0, const _Float16* p1) {
  v8h lo = *(const v8h*)p0;
  v8h hi = *(const v8h*)p1;
  v16h r;
#pragma unroll
  for (int i = 0; i < 8; ++i) { r[i] = lo[i]; r[i + 8] = hi[i]; }
  return r;
}

__global__ __launch_bounds__(256)
void dpi_fused_kernel(const float* __restrict__ X,
                      const float* __restrict__ Wa,
                      const float* __restrict__ Wsh,
                      const float* __restrict__ pIwA,
                      const float* __restrict__ pIwS,
                      const float* __restrict__ ImemIn,
                      const float* __restrict__ IampaIn,
                      const float* __restrict__ IshuntIn,
                      const float* __restrict__ refrIn,
                      float* __restrict__ out)
{
  __shared__ _Float16 Xs [64  * LDX];
  __shared__ _Float16 WsA[128 * LDX];
  __shared__ _Float16 WsS[128 * LDX];

  const int mBase = blockIdx.y * 64;
  const int nBase = blockIdx.x * 128;
  const int tid   = threadIdx.x;
  const int lane  = tid & 31;
  const int wave  = tid >> 5;
  const int wm    = wave >> 2;   // 0..1  -> M sub-tile
  const int wn    = wave & 3;    // 0..3  -> N sub-tile
  const int lh    = lane >> 4;   // lane half (0/1)
  const int l15   = lane & 15;

  v8f accA[2][2] = {};
  v8f accS[2][2] = {};

  for (int k0 = 0; k0 < N_IN; k0 += 32) {
    // ---- cooperative load + f32->f16 convert into LDS ----
#pragma unroll
    for (int q = 0; q < 2; ++q) {             // X tile: 64 x 32
      int i = tid + q * 256;
      int r = i >> 3, c4 = i & 7;
      size_t g = (size_t)(mBase + r) * N_IN + k0 + c4 * 4;
      float4 v = *(const float4*)(X + g);
      __builtin_prefetch(X + g + 32, 0, 0);   // next K tile -> global_prefetch
      v4h h = { (_Float16)v.x, (_Float16)v.y, (_Float16)v.z, (_Float16)v.w };
      *(v4h*)(Xs + r * LDX + c4 * 4) = h;
    }
#pragma unroll
    for (int q = 0; q < 4; ++q) {             // W tiles: 128 x 32 (x2), STE round
      int i = tid + q * 256;
      int n = i >> 3, c4 = i & 7;
      size_t g = (size_t)(nBase + n) * N_IN + k0 + c4 * 4;
      float4 va = *(const float4*)(Wa + g);
      v4h ha = { (_Float16)rintf(va.x), (_Float16)rintf(va.y),
                 (_Float16)rintf(va.z), (_Float16)rintf(va.w) };
      *(v4h*)(WsA + n * LDX + c4 * 4) = ha;
      float4 vs = *(const float4*)(Wsh + g);
      v4h hs = { (_Float16)rintf(vs.x), (_Float16)rintf(vs.y),
                 (_Float16)rintf(vs.z), (_Float16)rintf(vs.w) };
      *(v4h*)(WsS + n * LDX + c4 * 4) = hs;
      __builtin_prefetch(Wa + g + 32, 0, 0);
      __builtin_prefetch(Wsh + g + 32, 0, 0);
    }
    __syncthreads();

    // ---- fragment gathers (contiguous K per lane half) ----
    v16h a[2], ba[2], bs[2];
#pragma unroll
    for (int mt = 0; mt < 2; ++mt) {
      // A 16x32: lane-half 0 -> K 0..15, lane-half 1 -> K 16..31 (contiguous)
      const _Float16* base = Xs + (wm * 32 + mt * 16 + l15) * LDX;
      a[mt] = ld_frag(base + lh * 16, base + lh * 16 + 8);
    }
#pragma unroll
    for (int nt = 0; nt < 2; ++nt) {
      // B 32x16: lane-half 0 -> K 0..15, lane-half 1 -> K 16..31 (contiguous)
      const _Float16* bA = WsA + (wn * 32 + nt * 16 + l15) * LDX + lh * 16;
      ba[nt] = ld_frag(bA, bA + 8);
      const _Float16* bS = WsS + (wn * 32 + nt * 16 + l15) * LDX + lh * 16;
      bs[nt] = ld_frag(bS, bS + 8);
    }

    // ---- 8 WMMAs per wave per K step (2M x 2N x 2 GEMMs) ----
#pragma unroll
    for (int mt = 0; mt < 2; ++mt)
#pragma unroll
      for (int nt = 0; nt < 2; ++nt) {
        accA[mt][nt] = __builtin_amdgcn_wmma_f32_16x16x32_f16(
            false, a[mt], false, ba[nt], (short)0, accA[mt][nt], false, false);
        accS[mt][nt] = __builtin_amdgcn_wmma_f32_16x16x32_f16(
            false, a[mt], false, bs[nt], (short)0, accS[mt][nt], false, false);
      }
    __syncthreads();
  }

  // ---- fused DPI neuron epilogue ----
  const float IwA  = pIwA[0];
  const float IwS  = pIwS[0];
  const float cPow = powf(5e-14f, 0.5865102639296188f);   // I0^(1/(kappa+1))
  const size_t plane = (size_t)B_DIM * N_OUT;

  const float I0 = 5e-14f, Idc = 1e-12f, ItauMem = 1e-12f, IgainMem = 1e-12f;
  const float Ith = 1e-12f, IpfbN = 1e-12f, IpfbTh = 1e-12f;
  const float tauMem = 0.10638297872340426f;   // Ut/kappa*Cmem/Itau
  const float invTauA = 14.1f;                 // 1/tau_ampa = 1/tau_shunt
  const float Iahp = I0, Inmda = I0, beta = 0.05f, dtc = 1e-3f;

#pragma unroll
  for (int mt = 0; mt < 2; ++mt)
#pragma unroll
    for (int nt = 0; nt < 2; ++nt)
#pragma unroll
      for (int r = 0; r < 8; ++r) {
        int gm = mBase + wm * 32 + mt * 16 + r + lh * 8;   // C layout: VGPR r -> row, lane-half -> +8
        int gn = nBase + wn * 32 + nt * 16 + l15;          //           lane 0..15 -> col
        size_t idx = (size_t)gm * N_OUT + gn;
        float numA = accA[mt][nt][r];
        float numS = accS[mt][nt][r];

        float Imem   = __builtin_nontemporal_load(ImemIn   + idx);
        float Iampa  = __builtin_nontemporal_load(IampaIn  + idx);
        float Ishunt = __builtin_nontemporal_load(IshuntIn + idx);
        float refr   = __builtin_nontemporal_load(refrIn   + idx);

        float dIampa  = -Iampa  * invTauA;
        float IampaN  = Iampa  + IwA * numA;     // Igain_ampa/Itau_ampa == 1
        float dIshunt = -Ishunt * invTauA;
        float IshuntN = Ishunt + IwS * numS;

        float Iin = Idc + IampaN + Inmda - IshuntN;
        Iin = (refr <= 0.0f) ? Iin : 0.0f;
        Iin = fmaxf(Iin, I0);

        float Ifb = cPow * powf(Imem, 0.4134897360703812f)   // kappa/(kappa+1)
                    / (1.0f + expf(-IpfbN * (Imem - IpfbTh)));
        float f_imem = Ifb * 1e12f * (Imem + IgainMem);      // /Itau_mem

        float dImem = ((Iin - ItauMem - Iahp) - Imem         // alpha == 1
                       - (Iahp / I0) * beta * Imem + f_imem)
                      / (tauMem * (1.0f + IgainMem / Imem));
        float Imem2   = fmaxf(Imem   + dImem  * dtc, I0);
        float Iampa2  = fmaxf(IampaN + dIampa * dtc, I0);
        float Ishunt2 = fmaxf(IshuntN + dIshunt * dtc, I0);

        float spike = (Imem2 - Ith) > 0.0f ? 1.0f : 0.0f;
        float ImemO = (1.0f - spike) * Imem2 + spike * I0;
        float refr2 = fmaxf(refr - dtc, 0.0f);
        float refrO = (1.0f - spike) * refr2;                // refP == 0

        __builtin_nontemporal_store(spike,   out + idx);
        __builtin_nontemporal_store(ImemO,   out + plane     + idx);
        __builtin_nontemporal_store(Iampa2,  out + 2 * plane + idx);
        __builtin_nontemporal_store(Ishunt2, out + 3 * plane + idx);
        __builtin_nontemporal_store(refrO,   out + 4 * plane + idx);
      }
}

extern "C" void kernel_launch(void* const* d_in, const int* in_sizes, int n_in,
                              void* d_out, int out_size, void* d_ws, size_t ws_size,
                              hipStream_t stream) {
  (void)in_sizes; (void)n_in; (void)out_size; (void)d_ws; (void)ws_size;
  dim3 grid(N_OUT / 128, B_DIM / 64);   // 16 x 64 workgroups
  dpi_fused_kernel<<<grid, 256, 0, stream>>>(
      (const float*)d_in[0],  // X
      (const float*)d_in[1],  // W_ampa
      (const float*)d_in[2],  // W_shunt
      (const float*)d_in[3],  // Iw_ampa (scalar)
      (const float*)d_in[4],  // Iw_shunt (scalar)
      (const float*)d_in[5],  // Imem
      (const float*)d_in[6],  // Iampa
      (const float*)d_in[7],  // Ishunt
      (const float*)d_in[8],  // refractory
      (float*)d_out);
}
